// TreeAttention_70257075028607
// MI455X (gfx1250) — compile-verified, matrix operand order checked by reading, not run on hardware
//
#include <hip/hip_runtime.h>
#include <hip/hip_bf16.h>

typedef __attribute__((ext_vector_type(16))) _Float16 v16h;
typedef __attribute__((ext_vector_type(8)))  float    v8f;

namespace {
constexpr int N_ = 16, T_ = 2048, H_ = 64, S_ = 2048;
constexpr int KS = 64, INIT_W = 16;
constexpr int ZMAX = 2 * KS;     // max live candidates per level
constexpr int N_ITER = 7;        // log2(S / INIT_W)
constexpr float NEGV = -1e9f;
}

// Score 16 candidates (rows of A = gathered key rows, f32->f16) against the
// replicated-query B fragment. Two k=32 WMMAs cover H=64. Writes the 16
// scores (D column 0 per row) to sbuf[tb*16 .. tb*16+15].
__device__ __forceinline__ void score_tile16(const float* __restrict__ Kb,
                                             const int* posb, float* sbuf,
                                             int tb, int lane, int boff,
                                             v16h bf0, v16h bf1) {
  const float* kr = Kb + (size_t)posb[tb * 16 + (lane & 15)] * H_;
  v16h a0, a1;
#pragma unroll
  for (int i = 0; i < 8; ++i) {
    a0[i]     = (_Float16)kr[boff + i];            // k-chunk0, low half
    a0[i + 8] = (_Float16)kr[boff + 16 + i];       // k-chunk0, high half
    a1[i]     = (_Float16)kr[32 + boff + i];       // k-chunk1
    a1[i + 8] = (_Float16)kr[32 + boff + 16 + i];
  }
  v8f c = {};
  c = __builtin_amdgcn_wmma_f32_16x16x32_f16(false, a0, false, bf0,
                                             (short)0, c, false, false);
  c = __builtin_amdgcn_wmma_f32_16x16x32_f16(false, a1, false, bf1,
                                             (short)0, c, false, false);
  // D layout: VGPR r, lanes 0-15 -> M=r; lanes 16-31 -> M=8+r.
  if (lane == 0) {
#pragma unroll
    for (int r = 0; r < 8; ++r) sbuf[tb * 16 + r] = c[r];
  } else if (lane == 16) {
#pragma unroll
    for (int r = 0; r < 8; ++r) sbuf[tb * 16 + 8 + r] = c[r];
  }
}

__global__ __launch_bounds__(32)
void tree_attn_kernel(const float* __restrict__ Q, const float* __restrict__ Kk,
                      const float* __restrict__ Vv, float* __restrict__ Out) {
  __shared__ _Float16 q16[H_];
  __shared__ int   zb[2][ZMAX];   // double-buffered candidate cell ids
  __shared__ int   posb[ZMAX];
  __shared__ float sbuf[ZMAX];
  __shared__ int   validb[KS];
  __shared__ float pbuf[KS];

  const int t = blockIdx.x;
  const int n = blockIdx.y;
  const int lane = threadIdx.x;

  const float* qrow = Q + ((size_t)n * T_ + t) * H_;
  const float* Kb = Kk + (size_t)n * S_ * H_;
  const float* Vb = Vv + (size_t)n * S_ * H_;
  const float tsrc = (float)(S_ - T_ + 1 + t);  // causal source length

  // Query -> LDS as f16 (2 elems per lane); seed initial grid cells.
  q16[2 * lane]     = (_Float16)qrow[2 * lane];
  q16[2 * lane + 1] = (_Float16)qrow[2 * lane + 1];
  if (lane < INIT_W) zb[0][lane] = lane;
  __syncthreads();

  // Replicated-query B fragments (all 16 columns identical), built once.
  const int boff = (lane & 16) ? 8 : 0;
  v16h bf0, bf1;
#pragma unroll
  for (int i = 0; i < 8; ++i) {
    bf0[i]     = q16[boff + i];
    bf0[i + 8] = q16[boff + 16 + i];
    bf1[i]     = q16[32 + boff + i];
    bf1[i + 8] = q16[32 + boff + 16 + i];
  }

  int cur = INIT_W;
  int w = INIT_W;
  int cb = 0;

  // ---- hierarchical tree refinement ----
  for (int it = 0; it < N_ITER; ++it) {
    const int nb = cb ^ 1;
    const int cur2 = cur * 2;
    const int w2 = w * 2;

    // split: cell z -> children 2z, 2z+1
    for (int j = lane; j < cur; j += 32) {
      int z = zb[cb][j];
      zb[nb][2 * j]     = 2 * z;
      zb[nb][2 * j + 1] = 2 * z + 1;
    }
    __syncthreads();

    // representative key position per child cell (round-to-nearest-even)
    const float sc = tsrc / (float)w2;
    for (int j = lane; j < cur2; j += 32) {
      int p = (int)rintf((float)zb[nb][j] * sc);
      p = p < 0 ? 0 : (p > S_ - 1 ? S_ - 1 : p);
      posb[j] = p;
    }
    __syncthreads();

    // score all candidates, 16 per WMMA tile
    const int ntiles = cur2 >> 4;
    for (int tb = 0; tb < ntiles; ++tb)
      score_tile16(Kb, posb, sbuf, tb, lane, boff, bf0, bf1);
    __syncthreads();

    // keep top-KS by score (bitonic sort, descending; payload = z)
    if (cur2 > KS) {
      for (int k = 2; k <= cur2; k <<= 1) {
        for (int j = k >> 1; j > 0; j >>= 1) {
          for (int idx = lane; idx < cur2; idx += 32) {
            int ixj = idx ^ j;
            if (ixj > idx) {
              bool descReg = ((idx & k) == 0);
              float si = sbuf[idx], sj = sbuf[ixj];
              bool doswap = descReg ? (si < sj) : (si > sj);
              if (doswap) {
                sbuf[idx] = sj; sbuf[ixj] = si;
                int zi = zb[nb][idx]; zb[nb][idx] = zb[nb][ixj]; zb[nb][ixj] = zi;
              }
            }
          }
          __syncthreads();
        }
      }
      cur = KS;
    } else {
      cur = cur2;
    }
    cb = nb;
    w = w2;
  }

  // ---- final positions + causal mask (w == S now) ----
  const float scf = tsrc / (float)w;
  for (int j = lane; j < KS; j += 32) {
    int p = (int)rintf((float)zb[cb][j] * scf);
    p = p < 0 ? 0 : (p > S_ - 1 ? S_ - 1 : p);
    posb[j] = p;
    validb[j] = ((float)p < tsrc) ? 1 : 0;
  }
  __syncthreads();

  // final scores via WMMA
  for (int tb = 0; tb < KS / 16; ++tb)
    score_tile16(Kb, posb, sbuf, tb, lane, boff, bf0, bf1);
  __syncthreads();

  for (int j = lane; j < KS; j += 32)
    sbuf[j] = validb[j] ? sbuf[j] * 0.125f : NEGV;   // 1/sqrt(64)
  __syncthreads();

  // ---- softmax over 64 (2 scores per lane, wave-shuffle reductions) ----
  float s0 = sbuf[lane], s1 = sbuf[lane + 32];
  float m = fmaxf(s0, s1);
#pragma unroll
  for (int o = 16; o > 0; o >>= 1) m = fmaxf(m, __shfl_xor(m, o, 32));
  float e0 = __expf(s0 - m), e1 = __expf(s1 - m);
  if (!validb[lane])      e0 = 0.f;
  if (!validb[lane + 32]) e1 = 0.f;
  float ssum = e0 + e1;
#pragma unroll
  for (int o = 16; o > 0; o >>= 1) ssum += __shfl_xor(ssum, o, 32);
  float inv = 1.0f / ssum;  // z=0 -> pos=0 is always causal, sum > 0
  pbuf[lane]      = e0 * inv;
  pbuf[lane + 32] = e1 * inv;
  __syncthreads();

  // ---- out[h] = sum_z p[z] * V[pos_z][h]; lane owns h = 2*lane, 2*lane+1 ----
  float ax = 0.f, ay = 0.f;
  for (int z = 0; z < KS; ++z) {
    float pz = pbuf[z];
    const float* vr = Vb + (size_t)posb[z] * H_;
    ax = fmaf(pz, vr[2 * lane], ax);
    ay = fmaf(pz, vr[2 * lane + 1], ay);
  }
  float* orow = Out + ((size_t)n * T_ + t) * H_;
  orow[2 * lane]     = ax;
  orow[2 * lane + 1] = ay;
}

extern "C" void kernel_launch(void* const* d_in, const int* in_sizes, int n_in,
                              void* d_out, int out_size, void* d_ws, size_t ws_size,
                              hipStream_t stream) {
  (void)in_sizes; (void)n_in; (void)out_size; (void)d_ws; (void)ws_size;
  const float* Q = (const float*)d_in[0];
  const float* K = (const float*)d_in[1];
  const float* V = (const float*)d_in[2];
  float* O = (float*)d_out;
  dim3 grid(T_, N_);   // one wave32 block per query token
  tree_attn_kernel<<<grid, dim3(32, 1, 1), 0, stream>>>(Q, K, V, O);
}
